// HGT_16930761081097
// MI455X (gfx1250) — matrix operand order repaired and checked by reading, not compiled, not run on hardware
//
#include <hip/hip_runtime.h>
#include <hip/hip_bf16.h>
#include <math.h>

// ---------------------------------------------------------------------------
// HGT forward for MI455X (gfx1250, wave32, WMMA).
//
// Dense linear layers run through a templated WMMA GEMM kernel:
//   - fp32 in / fp32 out, f16 multiply via v_wmma_f32_16x16x32_f16
//   - 128-thread blocks = 4 wave32 waves
//   - Big config: 64x64 block tile, each wave owns a 32x32 output (2x2
//     accumulators) -> A/B fragments reused across 2 WMMAs each, i.e.
//     2 ds_load_b128 per WMMA instead of 4 (LDS issue-rate relief).
//   - Small config: 32x32 block tile for the per-head 32x32 rel transforms.
// Edge softmax/scatter phases are atomics-based VMEM kernels.
// ---------------------------------------------------------------------------

typedef __attribute__((ext_vector_type(16))) _Float16 v16h;
typedef __attribute__((ext_vector_type(8)))  _Float16 v8h;
typedef __attribute__((ext_vector_type(8)))  float    v8f;

#define HID   128
#define HEADS 4
#define DH    32

// Build a v16h WMMA fragment from two 16-byte LDS chunks at p+o1 / p+o2.
__device__ __forceinline__ v16h frag2(const _Float16* p, int o1, int o2) {
    const v8h lo = *(const v8h*)(p + o1);
    const v8h hi = *(const v8h*)(p + o2);
    v16h f;
    #pragma unroll
    for (int j = 0; j < 8; ++j) { f[j] = lo[j]; f[j + 8] = hi[j]; }
    return f;
}

// ---------------------------------------------------------------------------
// Templated WMMA GEMM: C = act(A[MxK] @ W[KxN] + bias)
// MB x NB = block tile; RT x CT = 16x16 tiles per wave (2x2 waves per block).
// lda/ldw/ldc = row strides (floats). blockIdx.z selects a "head" slice via
// Ahs/Whs/Chs element offsets. K in {32,128}, N multiple of NB.
// act: 0=none, 1=relu.
// ---------------------------------------------------------------------------
template<int MB, int NB, int RT, int CT>
__global__ __launch_bounds__(128)
void wmma_gemm_kernel(const float* __restrict__ A, int lda, long Ahs,
                      const float* __restrict__ W, int ldw, long Whs,
                      const float* __restrict__ bias,
                      float* __restrict__ C, int ldc, long Chs,
                      int M, int K, int N, int act)
{
    __shared__ __align__(16) _Float16 As[MB][136];   // As[r][k]
    __shared__ __align__(16) _Float16 Bs[NB][136];   // Bs[n][k] = W[k][col0+n]

    const int z = blockIdx.z;
    A += (long)z * Ahs;
    W += (long)z * Whs;
    C += (long)z * Chs;

    const int row0 = blockIdx.x * MB;
    const int col0 = blockIdx.y * NB;
    const int tid  = threadIdx.x;

    // Stage A tile (fp32 -> f16), float4 global loads, zero-pad rows beyond M.
    const int K4 = K >> 2;
    for (int idx = tid; idx < MB * K4; idx += 128) {
        const int r = idx / K4, k4 = idx - r * K4;
        float4 v = make_float4(0.f, 0.f, 0.f, 0.f);
        if (row0 + r < M) v = *(const float4*)&A[(long)(row0 + r) * lda + k4 * 4];
        _Float16* p = &As[r][k4 * 4];
        p[0] = (_Float16)v.x; p[1] = (_Float16)v.y;
        p[2] = (_Float16)v.z; p[3] = (_Float16)v.w;
    }
    // Stage W tile transposed (fp32 -> f16).
    for (int idx = tid; idx < NB * K; idx += 128) {
        const int k = idx / NB, n = idx - k * NB;
        Bs[n][k] = (_Float16)W[(long)k * ldw + (col0 + n)];
    }
    __syncthreads();

    const int wave = tid >> 5;
    const int lane = tid & 31;
    const int wr   = (wave & 1) * (RT * 16);   // wave row offset in block tile
    const int wc   = (wave >> 1) * (CT * 16);  // wave col offset in block tile
    const int m    = lane & 15;
    const int hi   = lane >> 4;                // lane group 0/1
    const int kbA  = hi * 8;                   // 16-bit A layout K split
    const int kbB  = hi * 8;                   // 16-bit B layout K split

    v8f acc[RT][CT] = {};
    for (int kk = 0; kk < K; kk += 32) {
        v16h af[RT], bf[CT];
        #pragma unroll
        for (int r = 0; r < RT; ++r)
            af[r] = frag2(&As[wr + r * 16 + m][kk], kbA, kbA + 16);
        #pragma unroll
        for (int c = 0; c < CT; ++c)
            bf[c] = frag2(&Bs[wc + c * 16 + m][kk], kbB, kbB + 16);
        #pragma unroll
        for (int r = 0; r < RT; ++r)
            #pragma unroll
            for (int c = 0; c < CT; ++c)
                acc[r][c] = __builtin_amdgcn_wmma_f32_16x16x32_f16(
                    /*neg_a=*/false, af[r], /*neg_b=*/false, bf[c],
                    /*c_mod=*/(short)0, acc[r][c],
                    /*reuse_a=*/false, /*reuse_b=*/false);
    }

    // C/D layout: vgpr i -> row (hi*8 + i), col (lane&15).
    #pragma unroll
    for (int c = 0; c < CT; ++c) {
        const int ccol = col0 + wc + c * 16 + m;
        const float bv = bias ? bias[ccol] : 0.0f;
        #pragma unroll
        for (int r = 0; r < RT; ++r) {
            #pragma unroll
            for (int i = 0; i < 8; ++i) {
                const int crow = row0 + wr + r * 16 + hi * 8 + i;
                if (crow < M) {
                    float v = acc[r][c][i] + bv;
                    if (act == 1) v = v > 0.0f ? v : 0.0f;
                    C[(long)crow * ldc + ccol] = v;
                }
            }
        }
    }
}

// ---------------------------------------------------------------------------
// Edge-softmax phases (atomics). Ordered-uint encoding for float atomicMax.
// ---------------------------------------------------------------------------
__device__ __forceinline__ unsigned enc_f32(float f) {
    unsigned u = __float_as_uint(f);
    return (u & 0x80000000u) ? ~u : (u | 0x80000000u);
}
__device__ __forceinline__ float dec_f32(unsigned u) {
    return (u & 0x80000000u) ? __uint_as_float(u & 0x7fffffffu)
                             : __uint_as_float(~u);
}

__global__ void edge_alpha_max_kernel(const float* __restrict__ qdst,
                                      const float* __restrict__ krel,
                                      const int*   __restrict__ ei, int E,
                                      const float* __restrict__ prel, float scale,
                                      float* __restrict__ alpha,
                                      unsigned* __restrict__ mbits)
{
    const int idx = blockIdx.x * blockDim.x + threadIdx.x;
    if (idx >= E * HEADS) return;
    const int e = idx >> 2, h = idx & 3;
    const int si = ei[e], di = ei[E + e];
    const float* qp = qdst + (long)di * HID + h * DH;
    const float* kp = krel + (long)si * HID + h * DH;
    float s = 0.0f;
    #pragma unroll
    for (int d = 0; d < DH; ++d) s += qp[d] * kp[d];
    s *= prel[h] * scale;
    alpha[idx] = s;
    atomicMax(&mbits[di * HEADS + h], enc_f32(s));
}

__global__ void edge_exp_sum_kernel(const int* __restrict__ ei, int E,
                                    float* __restrict__ alpha,
                                    const unsigned* __restrict__ mbits,
                                    float* __restrict__ ssum)
{
    const int idx = blockIdx.x * blockDim.x + threadIdx.x;
    if (idx >= E * HEADS) return;
    const int e = idx >> 2, h = idx & 3;
    const int di = ei[E + e];
    float m = dec_f32(mbits[di * HEADS + h]);
    if (!(m > -1e38f && m < 1e38f)) m = 0.0f;   // mirrors isfinite() clamp
    const float ev = expf(alpha[idx] - m);
    alpha[idx] = ev;
    atomicAdd(&ssum[di * HEADS + h], ev);
}

__global__ void edge_scatter_kernel(const int* __restrict__ ei, int E,
                                    const float* __restrict__ alpha,
                                    const float* __restrict__ ssum,
                                    const float* __restrict__ vrel,
                                    float* __restrict__ out)
{
    const int idx = blockIdx.x * blockDim.x + threadIdx.x;
    if (idx >= E * HID) return;
    const int e = idx >> 7, c = idx & 127, h = c >> 5;
    const int si = ei[e], di = ei[E + e];
    const float a = alpha[e * HEADS + h] / (ssum[di * HEADS + h] + 1e-16f);
    atomicAdd(&out[(long)di * HID + c], vrel[(long)si * HID + c] * a);
}

// ---------------------------------------------------------------------------
// Elementwise kernels
// ---------------------------------------------------------------------------
__global__ void gelu_kernel(const float* __restrict__ in,
                            float* __restrict__ out, long n)
{
    const long i = (long)blockIdx.x * blockDim.x + threadIdx.x;
    if (i < n) {
        const float x = in[i];
        out[i] = 0.5f * x * (1.0f + erff(x * 0.70710678118654752f));
    }
}

__global__ void skip_combine_kernel(const float* __restrict__ skip,
                                    const float* __restrict__ xold,
                                    float* __restrict__ io, long n)
{
    const long i = (long)blockIdx.x * blockDim.x + threadIdx.x;
    if (i < n) {
        const float sk = 1.0f / (1.0f + expf(-skip[0]));
        io[i] = sk * io[i] + (1.0f - sk) * xold[i];
    }
}

// ---------------------------------------------------------------------------
// Host orchestration.
//
// Input index map (setup_inputs() insertion order; params pytree flattened
// with JAX's sorted-dict-key convention):
//   0..3   : x_protocol, x_impression, x_medication, x_procedure
//   4..9   : ei_has, ei_suggests, ei_takes, ei_indicates, ei_usedby, ei_takenby
//   10..17 : in.{impression,medication,procedure,protocol}.{b,w}
//   18+54*l: layer l, keys sorted: a(8), a_rel(6), k(8), m_rel(6), p_rel(6),
//            q(8), skip(4), v(8); types sorted {impression,medication,
//            procedure,protocol}; rels sorted {has,indicates,suggests,
//            takenby,takes,usedby}
//   126,127: out.b, out.w
// ---------------------------------------------------------------------------
extern "C" void kernel_launch(void* const* d_in, const int* in_sizes, int n_in,
                              void* d_out, int out_size, void* d_ws, size_t ws_size,
                              hipStream_t stream)
{
    (void)n_in; (void)out_size; (void)ws_size;

    // Node types in NODE_TYPES order: protocol, impression, medication, procedure
    int ncnt[4], nbase[4];
    nbase[0] = 0;
    for (int t = 0; t < 4; ++t) {
        ncnt[t] = in_sizes[t] / HID;
        if (t > 0) nbase[t] = nbase[t - 1] + ncnt[t - 1];
    }
    const long NTOT = (long)nbase[3] + ncnt[3];          // 60000
    const int  E    = in_sizes[4] / 2;                   // 150000
    const int  sortpos[4] = {3, 0, 1, 2};                // type -> sorted param slot

    struct Rel { int src, dst, ei_idx, rs; };
    const Rel rels[6] = {
        {0, 1, 4, 0},   // has:       protocol  -> impression
        {0, 2, 5, 2},   // suggests:  protocol  -> medication
        {0, 3, 6, 4},   // takes:     protocol  -> procedure
        {1, 0, 7, 1},   // indicates: impression-> protocol
        {2, 0, 8, 5},   // usedby:    medication-> protocol
        {3, 0, 9, 3},   // takenby:   procedure -> protocol
    };

    auto P = [&](int i) { return (const float*)d_in[i]; };

    // Workspace layout (floats)
    float* ws = (float*)d_ws;
    const long FB = NTOT * HID;                          // 7,680,000 floats
    float* X0   = ws;
    float* X1   = ws + FB;
    float* Kb   = ws + 2 * FB;
    float* Qb   = ws + 3 * FB;
    float* Vb   = ws + 4 * FB;
    float* ATT  = ws + 5 * FB;
    float* GEL  = ws + 6 * FB;
    float* KREL = ws + 7 * FB;                           // max 20000*128
    float* VREL = KREL + 20000L * HID;
    float* ALPHA= VREL + 20000L * HID;                   // E*HEADS
    float* SSUM = ALPHA + (long)E * HEADS;               // 20000*HEADS
    unsigned* MB_ = (unsigned*)(SSUM + 20000L * HEADS);  // 20000*HEADS

    // Big GEMM: 64x64 block tile, waves own 32x32 (2x2 accumulators).
    auto gemm_big = [&](const float* A, const float* W, const float* bias,
                        float* C, int M, int N, int act) {
        dim3 grid((M + 63) / 64, N / 64, 1);
        wmma_gemm_kernel<64, 64, 2, 2><<<grid, dim3(128), 0, stream>>>(
            A, HID, 0, W, N, 0, bias, C, (N == 64 ? 64 : HID), 0, M, HID, N, act);
    };
    // Small per-head GEMM: [ns x 32] @ [32 x 32], 4 heads via grid.z.
    auto gemm_rel = [&](const float* A, const float* W, float* C, int M) {
        dim3 grid((M + 31) / 32, 1, HEADS);
        wmma_gemm_kernel<32, 32, 1, 1><<<grid, dim3(128), 0, stream>>>(
            A, HID, DH, W, DH, DH * DH, nullptr, C, HID, DH, M, DH, DH, 0);
    };

    const float scale = 0.17677669529663689f;            // 1/sqrt(32)

    // ---- input projection + ReLU ----
    for (int t = 0; t < 4; ++t) {
        const int sp = sortpos[t];
        gemm_big((const float*)d_in[t], P(10 + sp * 2 + 1), P(10 + sp * 2),
                 X0 + (long)nbase[t] * HID, ncnt[t], HID, /*relu*/1);
    }

    float* xcur = X0;
    float* xnxt = X1;

    for (int l = 0; l < 2; ++l) {
        const int lb = 18 + 54 * l;

        // K, Q, V projections per type
        for (int t = 0; t < 4; ++t) {
            const int sp = sortpos[t];
            const long ob = (long)nbase[t] * HID;
            gemm_big(xcur + ob, P(lb + 14 + sp * 2 + 1), P(lb + 14 + sp * 2),
                     Kb + ob, ncnt[t], HID, 0);
            gemm_big(xcur + ob, P(lb + 34 + sp * 2 + 1), P(lb + 34 + sp * 2),
                     Qb + ob, ncnt[t], HID, 0);
            gemm_big(xcur + ob, P(lb + 46 + sp * 2 + 1), P(lb + 46 + sp * 2),
                     Vb + ob, ncnt[t], HID, 0);
        }

        hipMemsetAsync(ATT, 0, FB * sizeof(float), stream);

        for (int ri = 0; ri < 6; ++ri) {
            const Rel& r = rels[ri];
            const int ns = ncnt[r.src], nd = ncnt[r.dst];
            const long sb = (long)nbase[r.src] * HID;
            const long db = (long)nbase[r.dst] * HID;

            gemm_rel(Kb + sb, P(lb + 8 + r.rs),  KREL, ns);
            gemm_rel(Vb + sb, P(lb + 22 + r.rs), VREL, ns);

            hipMemsetAsync(MB_,  0, (size_t)nd * HEADS * sizeof(unsigned), stream);
            hipMemsetAsync(SSUM, 0, (size_t)nd * HEADS * sizeof(float), stream);

            const int* ei = (const int*)d_in[r.ei_idx];
            const int eh = E * HEADS;
            edge_alpha_max_kernel<<<(eh + 255) / 256, 256, 0, stream>>>(
                Qb + db, KREL, ei, E, P(lb + 28 + r.rs), scale, ALPHA, MB_);
            edge_exp_sum_kernel<<<(eh + 255) / 256, 256, 0, stream>>>(
                ei, E, ALPHA, MB_, SSUM);
            const long ec = (long)E * HID;
            edge_scatter_kernel<<<(int)((ec + 255) / 256), 256, 0, stream>>>(
                ei, E, ALPHA, SSUM, VREL, ATT + db);
        }

        // GELU -> a-projection -> sigmoid-skip combine
        gelu_kernel<<<(int)((FB + 255) / 256), 256, 0, stream>>>(ATT, GEL, FB);
        for (int t = 0; t < 4; ++t) {
            const int sp = sortpos[t];
            const long ob = (long)nbase[t] * HID;
            gemm_big(GEL + ob, P(lb + 0 + sp * 2 + 1), P(lb + 0 + sp * 2),
                     xnxt + ob, ncnt[t], HID, 0);
        }
        for (int t = 0; t < 4; ++t) {
            const int sp = sortpos[t];
            const long ob = (long)nbase[t] * HID;
            const long n  = (long)ncnt[t] * HID;
            skip_combine_kernel<<<(int)((n + 255) / 256), 256, 0, stream>>>(
                P(lb + 42 + sp), xcur + ob, xnxt + ob, n);
        }
        float* tmp = xcur; xcur = xnxt; xnxt = tmp;
    }

    // ---- final output projection: protocol nodes -> [20000 x 64] ----
    gemm_big(xcur + (long)nbase[0] * HID, P(127), P(126),
             (float*)d_out, ncnt[0], 64, 0);
}
